// SC_MoEBlock_2018634629728
// MI455X (gfx1250) — compile-verified, matrix operand order checked by reading, not run on hardware
//
#include <hip/hip_runtime.h>
#include <hip/hip_bf16.h>
#include <math.h>

// ---------------- problem constants (match reference) ----------------
constexpr int Dm  = 1024;
constexpr int NH  = 16;
constexpr int NE  = 8;
constexpr int Bq  = 4;
constexpr int Sq  = 512;
constexpr int HID = 4 * Dm;
constexpr int Tt  = Bq * Sq;            // 2048 tokens per stream
constexpr long long TD = (long long)Tt * Dm;

// ---------------- input index map (setup_inputs dict order) ----------
enum {
  IN_XSWIN = 0, IN_XCONV = 1,
  IN_LN_SWIN_PRE_G = 2, IN_LN_SWIN_PRE_B = 3,
  IN_LN_CONV_PRE_G = 4, IN_LN_CONV_PRE_B = 5,
  IN_LN_SWIN_SELF_G = 6, IN_LN_SWIN_SELF_B = 7,
  IN_LN_CONV_SELF_G = 8, IN_LN_CONV_SELF_B = 9,
  IN_LN_MOE_G = 10, IN_LN_MOE_B = 11,
  IN_CS_WIN = 12, IN_CS_BIN = 13, IN_CS_WOUT = 14, IN_CS_BOUT = 15,
  IN_RW = 28, IN_RB = 29, IN_NW = 30, IN_NB = 31,
  IN_W1 = 32, IN_B1 = 33, IN_W2 = 34, IN_B2 = 35
};

// ---------------- bf16 helpers -------------------------------------
typedef __bf16 bf16_t;
typedef bf16_t v16bf __attribute__((ext_vector_type(16)));
typedef float  v8f   __attribute__((ext_vector_type(8)));

__device__ __forceinline__ unsigned short bfbits(float f) {
  unsigned u = __builtin_bit_cast(unsigned, f);
  u += 0x7FFFu + ((u >> 16) & 1u);               // RNE
  return (unsigned short)(u >> 16);
}
__device__ __forceinline__ bf16_t f2bf(float f) {
  return __builtin_bit_cast(bf16_t, bfbits(f));
}
__device__ __forceinline__ float bf2f(bf16_t h) {
  unsigned u = (unsigned)__builtin_bit_cast(unsigned short, h) << 16;
  return __builtin_bit_cast(float, u);
}

#define FLAG_ACC   1
#define FLAG_GELU  2
#define FLAG_OBF16 4
#define FLAG_VT    8

// LDS staging: two 12KB stages (A 128x32 bf16 = 8KB, B 64x32 bf16 = 4KB)
#define STAGE_BYTES 12288
#define LDS_BYTES   (2 * STAGE_BYTES)

// ---------------- optional Tensor Data Mover staging ---------------
#if __has_builtin(__builtin_amdgcn_tensor_load_to_lds)
#define HAVE_TDM 1
typedef unsigned int u32x4 __attribute__((ext_vector_type(4)));
typedef int i32x8 __attribute__((ext_vector_type(8)));
typedef int i32x4 __attribute__((ext_vector_type(4)));

// 2-D bf16 tile (32 elements wide x `rows` tall, row stride `strideEl`)
// global -> LDS offset `ldsOff`, contiguous rows of 64B in LDS.
__device__ __forceinline__ void tdm_load_tile(const bf16_t* g, unsigned ldsOff,
                                              int rows, int strideEl) {
  unsigned long long ga = (unsigned long long)(uintptr_t)g;
  u32x4 g0;
  g0[0] = 1u;                                           // count=1, user desc
  g0[1] = ldsOff;                                       // lds_addr (bytes)
  g0[2] = (unsigned)(ga & 0xffffffffu);                 // global_addr[31:0]
  g0[3] = (unsigned)((ga >> 32) & 0x1ffffffu)           // global_addr[56:32]
        | (2u << 30);                                   // type=2 ("image")
  i32x8 g1;
  g1[0] = (int)(1u << 16);                              // data_size=2B
  g1[1] = (int)(32u << 16);                             // tensor_dim0=32
  g1[2] = (int)(((unsigned)rows) << 16);                // tensor_dim1=rows
  g1[3] = (int)(32u << 16);                             // tile_dim0=32
  g1[4] = rows;                                         // tile_dim1=rows
  g1[5] = strideEl;                                     // tensor_dim0_stride
  g1[6] = 0; g1[7] = 0;
  i32x4 z4 = {0, 0, 0, 0};
  i32x8 z8 = {0, 0, 0, 0, 0, 0, 0, 0};
  __builtin_amdgcn_tensor_load_to_lds(g0, g1, z4, z4, z8, 0);
}
#else
#define HAVE_TDM 0
#endif

// ---------------- generic batched bf16 WMMA GEMM --------------------
// C[b,h][M,N] = alpha * A[b,h][M,K] x B[b,h][K,N] (+bias/resid/gelu/rowscale)
// A: bf16 row-major [M,K]; B: bf16 [N,K] row-major (always "NK").
// Block tile 128x64, 8 wave32s, each wave 32x32 (4 WMMA per K-step of 32).
// TDM path: double-buffered LDS stages; next tile pair is in flight while the
// current one feeds the WMMAs (s_wait_tensorcnt <= 2 retires the older pair).
struct GemmP {
  const bf16_t* A; const bf16_t* B; void* C;
  const float* bias; const float* resid; const float* rowScale;
  long long Ab, Ah, Bb, Bh, Cb, Ch;
  int lda, ldb, ldc, ldr;
  int M, N, K, hdiv, rsStride, flags;
  float alpha;
};

union Frag { uint4 u[2]; v16bf v; };

__global__ __launch_bounds__(256) void gemm_wmma_kernel(GemmP p) {
  extern __shared__ char smem[];
  const int z  = blockIdx.z;
  const int bb = z / p.hdiv, hh = z % p.hdiv;
  const bf16_t* A = p.A + (long long)bb * p.Ab + (long long)hh * p.Ah;
  const bf16_t* B = p.B + (long long)bb * p.Bb + (long long)hh * p.Bh;
  void* Cv;
  if (p.flags & (FLAG_OBF16 | FLAG_VT))
    Cv = (void*)((bf16_t*)p.C + (long long)bb * p.Cb + (long long)hh * p.Ch);
  else
    Cv = (void*)((float*)p.C + (long long)bb * p.Cb + (long long)hh * p.Ch);

  const int m0 = blockIdx.y * 128;
  const int n0 = blockIdx.x * 64;
  const int tid = threadIdx.x;
  const int lane = tid & 31, wave = tid >> 5;
  const int wm = wave & 3;        // wave rows  [wm*32, wm*32+32)
  const int wn = wave >> 2;       // wave cols  [wn*32, wn*32+32)
  const int lh = lane >> 4, ln16 = lane & 15;

  v8f acc00 = {}, acc01 = {}, acc10 = {}, acc11 = {};

#if HAVE_TDM
  // prologue: fill stage 0
  if (tid < 32) {
    tdm_load_tile(A + (long long)m0 * p.lda, 0u, 128, p.lda);
    tdm_load_tile(B + (long long)n0 * p.ldb, 8192u, 64, p.ldb);
  }
#endif

  for (int k0 = 0; k0 < p.K; k0 += 32) {
    const unsigned stg  = ((unsigned)(k0 >> 5) & 1u) * STAGE_BYTES;
    const char* sbase = smem + stg;
#if HAVE_TDM
    const bool more = (k0 + 32 < p.K);
    if (tid < 32) {                 // wave 0 drives the TDM pipeline
      if (more) {
        const unsigned nstg = stg ^ STAGE_BYTES;
        tdm_load_tile(A + (long long)m0 * p.lda + k0 + 32, nstg, 128, p.lda);
        tdm_load_tile(B + (long long)n0 * p.ldb + k0 + 32, nstg + 8192u, 64, p.ldb);
        __builtin_amdgcn_s_wait_tensorcnt(2);   // current stage landed
      } else {
        __builtin_amdgcn_s_wait_tensorcnt(0);   // drain
      }
    }
    __syncthreads();                 // stage `stg` visible to all waves
#else
    // manual fallback: copy current tile into this stage
#pragma unroll
    for (int i = 0; i < 2; ++i) {
      const int idx = i * 256 + tid;
      const int row = idx >> 2, c = idx & 3;
      *(uint4*)((char*)sbase + row * 64 + c * 16) =
          *(const uint4*)(A + (long long)(m0 + row) * p.lda + k0 + c * 8);
    }
    {
      const int row = tid >> 2, c = tid & 3;
      *(uint4*)((char*)sbase + 8192 + row * 64 + c * 16) =
          *(const uint4*)(B + (long long)(n0 + row) * p.ldb + k0 + c * 8);
    }
    if (k0 + 32 < p.K) {             // global_prefetch_b8
      __builtin_prefetch(A + (long long)(m0 + (tid & 127)) * p.lda + k0 + 32, 0, 1);
      __builtin_prefetch(B + (long long)(n0 + (tid & 63)) * p.ldb + k0 + 32, 0, 1);
    }
    __syncthreads();
#endif

    // Fragments per ISA 7.12.2: lane<16 holds K{0..7,16..23}, lane>=16 the rest.
    Frag a0, a1, b0, b1;
    {
      const char* sa = sbase + (wm * 32 + ln16) * 64 + lh * 16;
      a0.u[0] = *(const uint4*)(sa);
      a0.u[1] = *(const uint4*)(sa + 32);
      a1.u[0] = *(const uint4*)(sa + 16 * 64);
      a1.u[1] = *(const uint4*)(sa + 16 * 64 + 32);
      const char* sb = sbase + 8192 + (wn * 32 + ln16) * 64 + lh * 16;
      b0.u[0] = *(const uint4*)(sb);
      b0.u[1] = *(const uint4*)(sb + 32);
      b1.u[0] = *(const uint4*)(sb + 16 * 64);
      b1.u[1] = *(const uint4*)(sb + 16 * 64 + 32);
    }
    acc00 = __builtin_amdgcn_wmma_f32_16x16x32_bf16(false, a0.v, false, b0.v, (short)0, acc00, false, false);
    acc01 = __builtin_amdgcn_wmma_f32_16x16x32_bf16(false, a0.v, false, b1.v, (short)0, acc01, false, false);
    acc10 = __builtin_amdgcn_wmma_f32_16x16x32_bf16(false, a1.v, false, b0.v, (short)0, acc10, false, false);
    acc11 = __builtin_amdgcn_wmma_f32_16x16x32_bf16(false, a1.v, false, b1.v, (short)0, acc11, false, false);

    __syncthreads();                 // stage free before TDM re-targets it
  }

  // ---- epilogue ----
  auto store = [&](v8f& a, int i, int j) {
    const int col = n0 + wn * 32 + j * 16 + ln16;
    const float bias = p.bias ? p.bias[col] : 0.f;
#pragma unroll
    for (int r = 0; r < 8; ++r) {
      const int row = m0 + wm * 32 + i * 16 + lh * 8 + r;
      float v = a[r] * p.alpha + bias;
      if (p.flags & FLAG_GELU) v = 0.5f * v * (1.f + erff(v * 0.70710678118f));
      if (p.rowScale) v *= p.rowScale[(long long)row * p.rsStride];
      if (p.resid)    v += p.resid[(long long)row * p.ldr + col];
      if (p.flags & FLAG_VT) {
        // write V^T[b,h,d,s] (b=row/512, s=row%512, h=col/64, d=col%64)
        const int b = row >> 9, s = row & 511, h = col >> 6, d = col & 63;
        ((bf16_t*)Cv)[(((long long)(b * 16 + h) * 64 + d) << 9) + s] = f2bf(v);
      } else if (p.flags & FLAG_OBF16) {
        ((bf16_t*)Cv)[(long long)row * p.ldc + col] = f2bf(v);
      } else {
        float* cp = (float*)Cv + (long long)row * p.ldc + col;
        float o = v;
        if (p.flags & FLAG_ACC) o += *cp;
        *cp = o;
      }
    }
  };
  store(acc00, 0, 0); store(acc01, 0, 1); store(acc10, 1, 0); store(acc11, 1, 1);
}

// ---------------- f32 -> bf16 conversion (packed) -------------------
__global__ __launch_bounds__(256) void cvt_kernel(const float* __restrict__ x,
                                                  bf16_t* __restrict__ y, long long n4) {
  const long long i = (long long)blockIdx.x * 256 + threadIdx.x;
  if (i < n4) {
    const float4 v = ((const float4*)x)[i];
    uint2 o;
    o.x = (unsigned)bfbits(v.x) | ((unsigned)bfbits(v.y) << 16);
    o.y = (unsigned)bfbits(v.z) | ((unsigned)bfbits(v.w) << 16);
    ((uint2*)y)[i] = o;
  }
}

// transpose + convert: src [K,N] f32 -> dst [N,K] bf16 (batched over z)
__global__ __launch_bounds__(256) void cvtT_kernel(const float* __restrict__ src,
                                                   bf16_t* __restrict__ dst,
                                                   int K, int N) {
  __shared__ float tile[32][33];
  const long long sb = (long long)blockIdx.z * K * N;
  const int k0 = blockIdx.y * 32, n0 = blockIdx.x * 32;
  const int tx = threadIdx.x & 31, ty = threadIdx.x >> 5;
#pragma unroll
  for (int i = 0; i < 4; ++i) {
    const int k = ty * 4 + i;
    tile[k][tx] = src[sb + (long long)(k0 + k) * N + (n0 + tx)];
  }
  __syncthreads();
#pragma unroll
  for (int i = 0; i < 4; ++i) {
    const int n = ty * 4 + i;
    dst[sb + (long long)(n0 + n) * K + (k0 + tx)] = f2bf(tile[tx][n]);
  }
}

// ---------------- LayerNorm (f32 in, bf16 out) ----------------------
__global__ __launch_bounds__(256) void ln_kernel(const float* __restrict__ x,
                                                 const float* __restrict__ g,
                                                 const float* __restrict__ b,
                                                 bf16_t* __restrict__ y) {
  const int t = blockIdx.x, tid = threadIdx.x;
  const float* xr = x + (long long)t * 1024;
  __shared__ float red[256];
  float v0 = xr[tid], v1 = xr[tid + 256], v2 = xr[tid + 512], v3 = xr[tid + 768];
  red[tid] = v0 + v1 + v2 + v3; __syncthreads();
  for (int st = 128; st; st >>= 1) { if (tid < st) red[tid] += red[tid + st]; __syncthreads(); }
  const float mean = red[0] * (1.f / 1024.f); __syncthreads();
  const float d0 = v0 - mean, d1 = v1 - mean, d2 = v2 - mean, d3 = v3 - mean;
  red[tid] = d0 * d0 + d1 * d1 + d2 * d2 + d3 * d3; __syncthreads();
  for (int st = 128; st; st >>= 1) { if (tid < st) red[tid] += red[tid + st]; __syncthreads(); }
  const float inv = rsqrtf(red[0] * (1.f / 1024.f) + 1e-5f);
  bf16_t* yr = y + (long long)t * 1024;
  yr[tid]       = f2bf(d0 * inv * g[tid]       + b[tid]);
  yr[tid + 256] = f2bf(d1 * inv * g[tid + 256] + b[tid + 256]);
  yr[tid + 512] = f2bf(d2 * inv * g[tid + 512] + b[tid + 512]);
  yr[tid + 768] = f2bf(d3 * inv * g[tid + 768] + b[tid + 768]);
}

// ---------------- softmax rows of 512 (f32 in, bf16 out) ------------
__global__ __launch_bounds__(256) void softmax512_kernel(const float* __restrict__ x,
                                                         bf16_t* __restrict__ y) {
  const long long row = blockIdx.x;
  const float* r = x + row * 512;
  const int tid = threadIdx.x;
  __shared__ float red[256];
  const float a = r[tid], b2 = r[tid + 256];
  red[tid] = fmaxf(a, b2); __syncthreads();
  for (int st = 128; st; st >>= 1) { if (tid < st) red[tid] = fmaxf(red[tid], red[tid + st]); __syncthreads(); }
  const float mx = red[0]; __syncthreads();
  const float ea = __expf(a - mx), eb = __expf(b2 - mx);
  red[tid] = ea + eb; __syncthreads();
  for (int st = 128; st; st >>= 1) { if (tid < st) red[tid] += red[tid + st]; __syncthreads(); }
  const float inv = 1.f / red[0];
  bf16_t* w = y + row * 512;
  w[tid] = f2bf(ea * inv); w[tid + 256] = f2bf(eb * inv);
}

// ---------------- router (bf16 activations) -------------------------
__device__ __forceinline__ float gauss_hash(unsigned t, unsigned e) {
  unsigned s = t * 2654435761u ^ (e * 40503u + 0x9e3779b9u);
  s ^= s >> 16; s *= 0x85ebca6bu; s ^= s >> 13; s *= 0xc2b2ae35u; s ^= s >> 16;
  unsigned s2 = s * 0x27d4eb2du + 0x165667b1u; s2 ^= s2 >> 15; s2 *= 0x2c1b3c6du; s2 ^= s2 >> 12;
  const float u1 = (float)(s  & 0xFFFFFFu) * (1.f / 16777216.f) + 5.96e-8f;
  const float u2 = (float)(s2 & 0xFFFFFFu) * (1.f / 16777216.f);
  return sqrtf(-2.f * logf(u1)) * cosf(6.28318530718f * u2);
}

__global__ __launch_bounds__(256) void router_kernel(const bf16_t* __restrict__ x,
                                                     const float* __restrict__ rw,
                                                     const float* __restrict__ rb,
                                                     const float* __restrict__ nw,
                                                     const float* __restrict__ nb,
                                                     float* __restrict__ noisy) {
  const int t = blockIdx.x, tid = threadIdx.x;
  __shared__ float xs[1024];
  __shared__ float red[256];
  for (int i = tid; i < 1024; i += 256) xs[i] = bf2f(x[(long long)t * 1024 + i]);
  __syncthreads();
  const int g = tid >> 4, l = tid & 15;
  const float* w = (g < 8) ? (rw + g * 1024) : (nw + (g - 8) * 1024);
  float s = 0.f;
  for (int i = l; i < 1024; i += 16) s += xs[i] * w[i];
  red[tid] = s; __syncthreads();
  for (int st = 8; st; st >>= 1) { if (l < st) red[tid] += red[tid + st]; __syncthreads(); }
  if (tid < 8) {
    const float logit = red[tid * 16] + rb[tid];
    const float nlog  = red[(tid + 8) * 16] + nb[tid];
    const float sp = (nlog > 20.f) ? nlog : log1pf(__expf(nlog));
    noisy[(long long)t * 8 + tid] = logit + gauss_hash((unsigned)t, (unsigned)tid) * sp;
  }
}

__global__ __launch_bounds__(256) void topk_kernel(const float* __restrict__ noisy,
                                                   float* __restrict__ probs, int T) {
  const int t = blockIdx.x * blockDim.x + threadIdx.x;
  if (t >= T) return;
  float v[8];
#pragma unroll
  for (int e = 0; e < 8; ++e) v[e] = noisy[(long long)t * 8 + e];
  int i1 = 0;
#pragma unroll
  for (int e = 1; e < 8; ++e) if (v[e] > v[i1]) i1 = e;
  int i2 = (i1 == 0) ? 1 : 0;
#pragma unroll
  for (int e = 0; e < 8; ++e) if (e != i1 && v[e] > v[i2]) i2 = e;
  const float e2 = __expf(v[i2] - v[i1]);
  const float inv = 1.f / (1.f + e2);
#pragma unroll
  for (int e = 0; e < 8; ++e)
    probs[(long long)t * 8 + e] = (e == i1) ? inv : ((e == i2) ? e2 * inv : 0.f);
}

__global__ __launch_bounds__(256) void add_kernel(const float* __restrict__ a,
                                                  const float* __restrict__ b,
                                                  float* __restrict__ o, long long n) {
  const long long i = (long long)blockIdx.x * 256 + threadIdx.x;
  if (i < n) o[i] = a[i] + b[i];
}

// ---------------- host-side GEMM launcher ---------------------------
static void gemm(hipStream_t s,
                 const bf16_t* A, long long Ab, long long Ah, int lda,
                 const bf16_t* B, long long Bb, long long Bh, int ldb,
                 void* C, long long Cb, long long Ch, int ldc,
                 const float* bias, const float* resid, int ldr,
                 const float* rowScale, int rsStride,
                 float alpha, int M, int N, int K, int batches, int hdiv, int flags) {
  GemmP p;
  p.A = A; p.B = B; p.C = C; p.bias = bias; p.resid = resid; p.rowScale = rowScale;
  p.Ab = Ab; p.Ah = Ah; p.Bb = Bb; p.Bh = Bh; p.Cb = Cb; p.Ch = Ch;
  p.lda = lda; p.ldb = ldb; p.ldc = ldc; p.ldr = ldr;
  p.M = M; p.N = N; p.K = K; p.hdiv = hdiv; p.rsStride = rsStride;
  p.flags = flags; p.alpha = alpha;
  dim3 grid(N / 64, M / 128, batches);
  gemm_wmma_kernel<<<grid, 256, LDS_BYTES, s>>>(p);
}

// ---------------- full MHA ------------------------------------------
static void run_attention(hipStream_t s, const bf16_t* qinh, const bf16_t* kvinh,
                          const bf16_t* wInH, const float* b_in,
                          const bf16_t* wOutH, const float* b_out,
                          const float* resid, float* out,
                          bf16_t* projh, bf16_t* vth, float* scores,
                          bf16_t* probsh, bf16_t* ctxh) {
  const long long DD = (long long)Dm * Dm;
  // Q -> projh[:,0:1024], K -> projh[:,1024:2048]  (bf16 out)
  gemm(s, qinh, 0, 0, Dm, wInH, 0, 0, Dm, projh, 0, 0, 2 * Dm,
       b_in, nullptr, 0, nullptr, 0, 1.f, Tt, Dm, Dm, 1, 1, FLAG_OBF16);
  gemm(s, kvinh, 0, 0, Dm, wInH + DD, 0, 0, Dm, projh + Dm, 0, 0, 2 * Dm,
       b_in + Dm, nullptr, 0, nullptr, 0, 1.f, Tt, Dm, Dm, 1, 1, FLAG_OBF16);
  // V -> V^T[b,h,d,s] (bf16, transposed store so attn@V is an NK GEMM)
  gemm(s, kvinh, 0, 0, Dm, wInH + 2 * DD, 0, 0, Dm, vth, 0, 0, 0,
       b_in + 2 * Dm, nullptr, 0, nullptr, 0, 1.f, Tt, Dm, Dm, 1, 1, FLAG_VT);
  // scores[b,h] = (Q @ K^T)/8, f32 out
  gemm(s, projh, (long long)Sq * 2 * Dm, 64, 2 * Dm,
          projh + Dm, (long long)Sq * 2 * Dm, 64, 2 * Dm,
          scores, (long long)NH * Sq * Sq, (long long)Sq * Sq, Sq,
          nullptr, nullptr, 0, nullptr, 0,
          0.125f, Sq, Sq, 64, Bq * NH, NH, 0);
  softmax512_kernel<<<Bq * NH * Sq, 256, 0, s>>>(scores, probsh);
  // ctx[b,s,h,:] = attn @ V, bf16 out
  gemm(s, probsh, (long long)NH * Sq * Sq, (long long)Sq * Sq, Sq,
          vth, (long long)NH * 64 * Sq, (long long)64 * Sq, Sq,
          ctxh, (long long)Sq * Dm, 64, Dm,
          nullptr, nullptr, 0, nullptr, 0,
          1.f, Sq, 64, Sq, Bq * NH, NH, FLAG_OBF16);
  // out = ctx @ Wout^T + bout + resid (f32 out)
  gemm(s, ctxh, 0, 0, Dm, wOutH, 0, 0, Dm, out, 0, 0, Dm,
       b_out, resid, Dm, nullptr, 0, 1.f, Tt, Dm, Dm, 1, 1, 0);
}

// ---------------- entry point ---------------------------------------
extern "C" void kernel_launch(void* const* d_in, const int* in_sizes, int n_in,
                              void* d_out, int out_size, void* d_ws, size_t ws_size,
                              hipStream_t stream) {
  (void)in_sizes; (void)n_in; (void)out_size; (void)ws_size;
  const float* x_swin = (const float*)d_in[IN_XSWIN];
  const float* x_conv = (const float*)d_in[IN_XCONV];
  auto P = [&](int i) { return (const float*)d_in[i]; };

  // ---- workspace layout (256B aligned chunks) ----
  char* base = (char*)d_ws; size_t off = 0;
  auto alloc = [&](size_t bytes) -> void* {
    void* p = base + off; off += (bytes + 255) & ~(size_t)255; return p;
  };
  float*  xs     = (float*)alloc(TD * 4);
  float*  xc     = (float*)alloc(TD * 4);
  float*  xsum   = (float*)alloc(TD * 4);
  float*  moe    = (float*)alloc(TD * 4);
  float*  scores = (float*)alloc(8 * TD * 4);             // B*H*S*S
  float*  noisy  = (float*)alloc((long long)Tt * 8 * 4);
  float*  probs  = (float*)alloc((long long)Tt * 8 * 4);
  bf16_t* xn1h   = (bf16_t*)alloc(TD * 2);
  bf16_t* xn2h   = (bf16_t*)alloc(TD * 2);
  bf16_t* projh  = (bf16_t*)alloc(2 * TD * 2);            // [T, 2D] Q|K
  bf16_t* vth    = (bf16_t*)alloc(TD * 2);                // [B,H,64,S]
  bf16_t* probsh = (bf16_t*)alloc(8 * TD * 2);
  bf16_t* ctxh   = (bf16_t*)alloc(TD * 2);
  bf16_t* hbufh  = (bf16_t*)alloc(4 * TD * 2);            // [T, HID]
  bf16_t* wInH[4]; bf16_t* wOutH[4];
  for (int a = 0; a < 4; ++a) {
    wInH[a]  = (bf16_t*)alloc((long long)3 * Dm * Dm * 2);
    wOutH[a] = (bf16_t*)alloc((long long)Dm * Dm * 2);
  }
  bf16_t* w1t = (bf16_t*)alloc((long long)NE * Dm * HID * 2);  // [E, HID, D]
  bf16_t* w2t = (bf16_t*)alloc((long long)NE * HID * Dm * 2);  // [E, D, HID]

  // ---- one-shot weight conversion to bf16 (also transposes MoE) ----
  for (int a = 0; a < 4; ++a) {
    const long long nIn4  = (long long)3 * Dm * Dm / 4;
    const long long nOut4 = (long long)Dm * Dm / 4;
    cvt_kernel<<<(unsigned)(nIn4 / 256), 256, 0, stream>>>(P(IN_CS_WIN + 4 * a), wInH[a], nIn4);
    cvt_kernel<<<(unsigned)(nOut4 / 256), 256, 0, stream>>>(P(IN_CS_WOUT + 4 * a), wOutH[a], nOut4);
  }
  { dim3 g(HID / 32, Dm / 32, NE); cvtT_kernel<<<g, 256, 0, stream>>>(P(IN_W1), w1t, Dm, HID); }
  { dim3 g(Dm / 32, HID / 32, NE); cvtT_kernel<<<g, 256, 0, stream>>>(P(IN_W2), w2t, HID, Dm); }

  // ---- pre-norms on original inputs ----
  ln_kernel<<<Tt, 256, 0, stream>>>(x_swin, P(IN_LN_SWIN_PRE_G), P(IN_LN_SWIN_PRE_B), xn1h);
  ln_kernel<<<Tt, 256, 0, stream>>>(x_conv, P(IN_LN_CONV_PRE_G), P(IN_LN_CONV_PRE_B), xn2h);

  // ---- cross attentions ----
  run_attention(stream, xn1h, xn2h, wInH[0], P(IN_CS_BIN), wOutH[0], P(IN_CS_BIN + 2),
                x_swin, xs, projh, vth, scores, probsh, ctxh);
  run_attention(stream, xn2h, xn1h, wInH[1], P(IN_CS_BIN + 4), wOutH[1], P(IN_CS_BIN + 6),
                x_conv, xc, projh, vth, scores, probsh, ctxh);

  // ---- self attentions ----
  ln_kernel<<<Tt, 256, 0, stream>>>(xs, P(IN_LN_SWIN_SELF_G), P(IN_LN_SWIN_SELF_B), xn1h);
  run_attention(stream, xn1h, xn1h, wInH[2], P(IN_CS_BIN + 8), wOutH[2], P(IN_CS_BIN + 10),
                xs, xs, projh, vth, scores, probsh, ctxh);
  ln_kernel<<<Tt, 256, 0, stream>>>(xc, P(IN_LN_CONV_SELF_G), P(IN_LN_CONV_SELF_B), xn1h);
  run_attention(stream, xn1h, xn1h, wInH[3], P(IN_CS_BIN + 12), wOutH[3], P(IN_CS_BIN + 14),
                xc, xc, projh, vth, scores, probsh, ctxh);

  // ---- x = xc + xs ----
  add_kernel<<<(unsigned)(TD / 256), 256, 0, stream>>>(xc, xs, xsum, TD);

  // ---- MoE ----
  ln_kernel<<<Tt, 256, 0, stream>>>(xsum, P(IN_LN_MOE_G), P(IN_LN_MOE_B), xn1h);
  router_kernel<<<Tt, 256, 0, stream>>>(xn1h, P(IN_RW), P(IN_RB), P(IN_NW), P(IN_NB), noisy);
  topk_kernel<<<Tt / 256, 256, 0, stream>>>(noisy, probs, Tt);

  for (int e = 0; e < NE; ++e) {
    const bf16_t* w1e = w1t + (long long)e * HID * Dm;   // [HID, D] NK
    const bf16_t* w2e = w2t + (long long)e * Dm * HID;   // [D, HID] NK
    const float*  b1e = P(IN_B1) + (long long)e * HID;
    const float*  b2e = P(IN_B2) + (long long)e * Dm;
    gemm(stream, xn1h, 0, 0, Dm, w1e, 0, 0, Dm, hbufh, 0, 0, HID,
         b1e, nullptr, 0, nullptr, 0, 1.f, Tt, HID, Dm, 1, 1, FLAG_GELU | FLAG_OBF16);
    gemm(stream, hbufh, 0, 0, HID, w2e, 0, 0, HID, moe, 0, 0, Dm,
         b2e, nullptr, 0, probs + e, NE, 1.f, Tt, Dm, HID, 1, 1, e ? FLAG_ACC : 0);
  }

  // ---- out = x + moe(ln(x)) ----
  add_kernel<<<(unsigned)(TD / 256), 256, 0, stream>>>(xsum, moe, (float*)d_out, TD);
}
